// MultiHeadAttention_68478958567826
// MI455X (gfx1250) — compile-verified, matrix operand order checked by reading, not run on hardware
//
#include <hip/hip_runtime.h>
#include <hip/hip_bf16.h>

// ---------------------------------------------------------------------------
// Fused windowed multi-head attention for MI455X (gfx1250), wave32 + WMMA.
//   B=32, W=256 -> 8192 windows; L=49 (pad to 64), D=192, H=6, hd=32.
//   One workgroup per window; QKV proj, attention, out proj all in LDS.
//   GEMMs in bf16 via v_wmma_f32_16x16x32_bf16 (f32 accumulate).
//   Register-blocked A-operands: each wave owns one m-tile, caches its 6
//   k-step A tiles in VGPRs and sweeps 6 n-tiles of B streamed from L2.
// ---------------------------------------------------------------------------

typedef __attribute__((ext_vector_type(16))) __bf16 v16bf;
typedef __attribute__((ext_vector_type(8)))  __bf16 v8bf;
typedef __attribute__((ext_vector_type(4)))  __bf16 v4bf;
typedef __attribute__((ext_vector_type(8)))  float  v8f;

#define NWIN   8192
#define LQ     49
#define LPAD   64
#define DIM    192
#define NHEAD  6
#define HDIM   32
#define WELEM  (DIM*DIM)

__device__ __constant__ float c_slopes[NHEAD] = {
    1.0f, 0.63095734448019325f, 0.39810717055349726f,
    0.25118864315095796f, 0.15848931924611134f, 0.1f};

__device__ __forceinline__ v16bf cat8(v8bf lo, v8bf hi) {
    return __builtin_shufflevector(lo, hi, 0,1,2,3,4,5,6,7,8,9,10,11,12,13,14,15);
}

// Load a 16x32 bf16 WMMA operand tile (A, or B when the contraction dim is
// contiguous per row/column in memory).
//  lane 0-15 : row r0+lane,    k = k0+{0..7} and k0+{16..23}
//  lane16-31 : row r0+lane-16, k = k0+{8..15} and k0+{24..31}
__device__ __forceinline__ v16bf load_tile(const __bf16* base, int stride,
                                           int r0, int k0, int lane) {
    int r  = r0 + (lane & 15);
    int kb = k0 + ((lane & 16) ? 8 : 0);
    const __bf16* p = base + r * stride + kb;
    v8bf lo = *(const v8bf*)(p);
    v8bf hi = *(const v8bf*)(p + 16);
    return cat8(lo, hi);
}

__device__ __forceinline__ v8f wmma_bf16(v16bf a, v16bf b, v8f c) {
    return __builtin_amdgcn_wmma_f32_16x16x32_bf16(false, a, false, b,
                                                   (short)0, c, false, false);
}

// ---------------------------------------------------------------------------
// Weight fp32 -> bf16 conversion (runs once; result lives in L2 thereafter).
// ---------------------------------------------------------------------------
__global__ void convert_weights_kernel(const float* __restrict__ wq,
                                       const float* __restrict__ wk,
                                       const float* __restrict__ wv,
                                       const float* __restrict__ wo,
                                       __bf16* __restrict__ dst) {
    int i = blockIdx.x * blockDim.x + threadIdx.x;
    if (i < WELEM) {
        dst[0*WELEM + i] = (__bf16)wq[i];
        dst[1*WELEM + i] = (__bf16)wk[i];
        dst[2*WELEM + i] = (__bf16)wv[i];
        dst[3*WELEM + i] = (__bf16)wo[i];
    }
}

// ---------------------------------------------------------------------------
// Fused attention: one block per (b,w) window, 256 threads = 8 wave32s.
// ---------------------------------------------------------------------------
__global__ __launch_bounds__(256)
void fused_window_attn_kernel(const float* __restrict__ xq,
                              const float* __restrict__ xkv,
                              const unsigned char* __restrict__ mask,
                              const __bf16* __restrict__ wbf,  // Wq,Wk,Wv,Wo bf16
                              const float* __restrict__ bq,
                              const float* __restrict__ bk,
                              const float* __restrict__ bv,
                              const float* __restrict__ bo,
                              float* __restrict__ out) {
    // LDS: 5 x 24576 B = 120 KB  (2 workgroups per 320 KB WGP)
    __shared__ __align__(16) __bf16        sQ [LPAD * DIM];     // Q  bf16 row-major
    __shared__ __align__(16) __bf16        sK [LPAD * DIM];     // K  bf16 row-major
    __shared__ __align__(16) __bf16        sVt[DIM * LPAD];     // V^T bf16 (d-major)
    __shared__ __align__(16) __bf16        sA [LPAD * DIM];     // Xq  -> attn (bf16)
    __shared__ __align__(16) unsigned char sB [LPAD * DIM * 2]; // Xkv -> S(f32)+P(bf16)

    __bf16* sXkv = (__bf16*)sB;
    float*  sS   = (float*)sB;                          // 64x64 f32 logits
    __bf16* sP   = (__bf16*)(sB + LPAD * LPAD * 4);     // 64x64 bf16 probs

    const int tid  = threadIdx.x;
    const int lane = tid & 31;
    const int wave = tid >> 5;
    const size_t bw = blockIdx.x;

    const float* xqB  = xq  + bw * (size_t)(LQ * DIM);
    const float* xkvB = xkv + bw * (size_t)(LQ * DIM);

    const __bf16* Wq = wbf + 0*WELEM;
    const __bf16* Wk = wbf + 1*WELEM;
    const __bf16* Wv = wbf + 2*WELEM;
    const __bf16* Wo = wbf + 3*WELEM;

    // ---- phase 0: stage x_q / x_kv into LDS as bf16 (pad rows 49..63 = 0) ----
    for (int i = tid; i < LPAD * (DIM / 4); i += 256) {
        int row = i / (DIM / 4);
        int c4  = i % (DIM / 4);
        float4 a = {0.f, 0.f, 0.f, 0.f};
        float4 b = {0.f, 0.f, 0.f, 0.f};
        if (row < LQ) {
            a = ((const float4*)xqB )[row * (DIM / 4) + c4];
            b = ((const float4*)xkvB)[row * (DIM / 4) + c4];
        }
        v4bf pa = {(__bf16)a.x, (__bf16)a.y, (__bf16)a.z, (__bf16)a.w};
        v4bf pb = {(__bf16)b.x, (__bf16)b.y, (__bf16)b.z, (__bf16)b.w};
        *(v4bf*)(sA   + row * DIM + c4 * 4) = pa;
        *(v4bf*)(sXkv + row * DIM + c4 * 4) = pb;
    }
    __syncthreads();

    // Wave work split for the D x D GEMMs: wave owns m-tile (wave>>1) and
    // n-tiles [nh*6, nh*6+6) with nh = wave&1. A tiles cached in registers.
    const int mt = wave >> 1;
    const int nh = wave & 1;
    const int m0 = mt * 16;
    const int nC = (lane & 15);                 // C-tile column within tile
    const int mb = m0 + ((lane & 16) ? 8 : 0);  // C-tile first row for this lane

    // ---- phase 1a: Q = Xq*Wq^T + bq ----
    {
        v16bf aT[6];
        #pragma unroll
        for (int kk = 0; kk < 6; ++kk) aT[kk] = load_tile(sA, DIM, m0, kk * 32, lane);
        for (int nt = 0; nt < 6; ++nt) {
            int n0 = (nh * 6 + nt) * 16;
            v8f acc = {};
            #pragma unroll
            for (int kk = 0; kk < 6; ++kk)
                acc = wmma_bf16(aT[kk], load_tile(Wq, DIM, n0, kk * 32, lane), acc);
            int n = n0 + nC;
            float bb = bq[n];
            #pragma unroll
            for (int j = 0; j < 8; ++j) sQ[(mb + j) * DIM + n] = (__bf16)(acc[j] + bb);
        }
    }
    // ---- phase 1b: K and V share the cached Xkv A tiles ----
    {
        v16bf aT[6];
        #pragma unroll
        for (int kk = 0; kk < 6; ++kk) aT[kk] = load_tile(sXkv, DIM, m0, kk * 32, lane);
        for (int nt = 0; nt < 6; ++nt) {
            int n0 = (nh * 6 + nt) * 16;
            int n = n0 + nC;
            v8f acc = {};
            #pragma unroll
            for (int kk = 0; kk < 6; ++kk)
                acc = wmma_bf16(aT[kk], load_tile(Wk, DIM, n0, kk * 32, lane), acc);
            float bb = bk[n];
            #pragma unroll
            for (int j = 0; j < 8; ++j) sK[(mb + j) * DIM + n] = (__bf16)(acc[j] + bb);

            v8f accv = {};
            #pragma unroll
            for (int kk = 0; kk < 6; ++kk)
                accv = wmma_bf16(aT[kk], load_tile(Wv, DIM, n0, kk * 32, lane), accv);
            float bv_ = bv[n];
            #pragma unroll
            for (int j = 0; j < 8; ++j)                 // V stored transposed
                sVt[n * LPAD + (mb + j)] = (__bf16)(accv[j] + bv_);
        }
    }
    __syncthreads();   // sA (Xq) and sXkv now dead -> reused as attn / S+P

    const float scale = 0.17677669529663687f;  // 1/sqrt(32)

    // ---- phase 2: per-head attention ----
    for (int h = 0; h < NHEAD; ++h) {
        // logits S = Qh * Kh^T : 4x4 tiles, K-dim = hd = 32 -> single WMMA each
        for (int t = wave; t < 16; t += 8) {
            int lm0 = (t >> 2) * 16, ln0 = (t & 3) * 16;
            v8f acc = {};
            acc = wmma_bf16(load_tile(sQ, DIM, lm0, h * HDIM, lane),
                            load_tile(sK, DIM, ln0, h * HDIM, lane), acc);
            int n = ln0 + nC;
            int lmb = lm0 + ((lane & 16) ? 8 : 0);
            #pragma unroll
            for (int j = 0; j < 8; ++j) sS[(lmb + j) * LPAD + n] = acc[j];
        }
        __syncthreads();

        // softmax with ALiBi-Manhattan bias + boolean mask (one row per thread)
        if (tid < LPAD) {
            int q = tid;
            if (q < LQ) {
                int qi = q / 7, qj = q % 7;
                float slope = c_slopes[h];
                const unsigned char* mrow = mask + bw * (size_t)(LQ * LQ) + q * LQ;
                float mx = -3.0e38f;
                for (int k = 0; k < LQ; ++k) {
                    int ki = k / 7, kj = k % 7;
                    int di = qi - ki; di = di < 0 ? -di : di;
                    int dj = qj - kj; dj = dj < 0 ? -dj : dj;
                    float s = sS[q * LPAD + k] * scale - slope * (float)(di + dj);
                    if (!mrow[k]) s = -__builtin_inff();
                    sS[q * LPAD + k] = s;
                    mx = fmaxf(mx, s);
                }
                float sum = 0.f;
                for (int k = 0; k < LQ; ++k) {
                    float e = __expf(sS[q * LPAD + k] - mx);
                    sS[q * LPAD + k] = e;
                    sum += e;
                }
                float inv = 1.0f / sum;
                for (int k = 0; k < LQ; ++k)
                    sP[q * LPAD + k] = (__bf16)(sS[q * LPAD + k] * inv);
                for (int k = LQ; k < LPAD; ++k) sP[q * LPAD + k] = (__bf16)0.f;
            } else {
                for (int k = 0; k < LPAD; ++k) sP[q * LPAD + k] = (__bf16)0.f;
            }
        }
        __syncthreads();

        // attn_h = P * Vh : 4 m-tiles x 2 n-tiles, K = 64 (2 WMMA steps)
        {
            int t = wave;                      // 8 tiles, exactly one per wave
            int pm0 = (t >> 1) * 16, pn0 = (t & 1) * 16;
            v8f acc = {};
            #pragma unroll
            for (int k0 = 0; k0 < LPAD; k0 += 32)
                acc = wmma_bf16(load_tile(sP,  LPAD, pm0, k0, lane),
                                load_tile(sVt, LPAD, h * HDIM + pn0, k0, lane), acc);
            int n = h * HDIM + pn0 + nC;
            int pmb = pm0 + ((lane & 16) ? 8 : 0);
            #pragma unroll
            for (int j = 0; j < 8; ++j) sA[(pmb + j) * DIM + n] = (__bf16)acc[j];
        }
        __syncthreads();   // protect sS/sP for next head
    }

    // ---- phase 3: out = attn * Wo^T + bo, stored straight to HBM ----
    float* outB = out + bw * (size_t)(LQ * DIM);
    {
        v16bf aT[6];
        #pragma unroll
        for (int kk = 0; kk < 6; ++kk) aT[kk] = load_tile(sA, DIM, m0, kk * 32, lane);
        for (int nt = 0; nt < 6; ++nt) {
            int n0 = (nh * 6 + nt) * 16;
            v8f acc = {};
            #pragma unroll
            for (int kk = 0; kk < 6; ++kk)
                acc = wmma_bf16(aT[kk], load_tile(Wo, DIM, n0, kk * 32, lane), acc);
            int n = n0 + nC;
            float bb = bo[n];
            #pragma unroll
            for (int j = 0; j < 8; ++j) {
                int m = mb + j;
                if (m < LQ) outB[m * DIM + n] = acc[j] + bb;
            }
        }
    }
}

// ---------------------------------------------------------------------------
extern "C" void kernel_launch(void* const* d_in, const int* in_sizes, int n_in,
                              void* d_out, int out_size, void* d_ws, size_t ws_size,
                              hipStream_t stream) {
    const float*         xq   = (const float*)d_in[0];
    const float*         xkv  = (const float*)d_in[1];
    const unsigned char* mask = (const unsigned char*)d_in[2];
    const float*         Wq   = (const float*)d_in[3];
    const float*         bq   = (const float*)d_in[4];
    const float*         Wk   = (const float*)d_in[5];
    const float*         bk   = (const float*)d_in[6];
    const float*         Wv   = (const float*)d_in[7];
    const float*         bv   = (const float*)d_in[8];
    const float*         Wo   = (const float*)d_in[9];
    const float*         bo   = (const float*)d_in[10];
    float*               out  = (float*)d_out;

    __bf16* wbf = (__bf16*)d_ws;   // 4 * 192*192 bf16 = 288 KB

    convert_weights_kernel<<<(WELEM + 255) / 256, 256, 0, stream>>>(Wq, Wk, Wv, Wo, wbf);
    fused_window_attn_kernel<<<NWIN, 256, 0, stream>>>(xq, xkv, mask, wbf,
                                                       bq, bk, bv, bo, out);
}
